// Net_13769665151070
// MI455X (gfx1250) — matrix operand order repaired and checked by reading, not compiled, not run on hardware
//
#include <hip/hip_runtime.h>
#include <hip/hip_bf16.h>

// ---------------- problem constants ----------------
#define TT   512        // sequence length
#define BB   128        // batch
#define DD   256        // hidden size per direction
#define GG   1024       // 4*D gate width
#define KK   18         // tag count
#define KKP  32         // padded tag count (2 N-tiles)

typedef unsigned short u16;
typedef __attribute__((ext_vector_type(16))) __bf16 v16bf;
typedef __attribute__((ext_vector_type(8)))  __bf16 v8bf;
typedef __attribute__((ext_vector_type(8)))  float  v8f;

// ---------------- workspace layout (bytes) ----------------
constexpr size_t OFF_X    = 0;                                   // u16 [T*B*D]
constexpr size_t SZ_X     = (size_t)TT*BB*DD*2;
constexpr size_t OFF_WCAT = OFF_X + SZ_X;                        // u16 [2][G][512] (k<256: W_ih, k>=256: W_hh)
constexpr size_t SZ_WCAT  = 2ull*GG*512*2;
constexpr size_t OFF_BIAS = OFF_WCAT + SZ_WCAT;                  // f32 [2][G]
constexpr size_t SZ_BIAS  = 2ull*GG*4;
constexpr size_t OFF_FCW  = OFF_BIAS + SZ_BIAS;                  // u16 [32][512] padded fc_w
constexpr size_t SZ_FCW   = (size_t)KKP*512*2;
constexpr size_t OFF_HALL = OFF_FCW + SZ_FCW;                    // u16 [2][T][B][D] (dir1 in reversed step order)
constexpr size_t SZ_HALL  = 2ull*TT*BB*DD*2;
constexpr size_t OFF_FEAT = OFF_HALL + SZ_HALL;                  // f32 [T][B][18]
constexpr size_t SZ_FEAT  = (size_t)TT*BB*KK*4;
constexpr size_t OFF_LLH  = OFF_FEAT + SZ_FEAT;                  // f32 [128]

// ---------------- helpers ----------------
__device__ __forceinline__ u16 f2bf(float f) {
  unsigned u = __builtin_bit_cast(unsigned, f);
  unsigned r = u + 0x7FFFu + ((u >> 16) & 1u);   // round-to-nearest-even
  return (u16)(r >> 16);
}

// A fragment (16x32 bf16): lane holds row (lane&15), 16 contiguous K values [c0..c0+15]
__device__ __forceinline__ v16bf ldA(const u16* p) {
  v8bf lo = *reinterpret_cast<const v8bf*>(p);
  v8bf hi = *reinterpret_cast<const v8bf*>(p + 8);
  return __builtin_shufflevector(lo, hi, 0,1,2,3,4,5,6,7,8,9,10,11,12,13,14,15);
}
// B fragment (32x16 bf16): lane holds column (lane&15), 16 contiguous K values
__device__ __forceinline__ v16bf ldB(const u16* p) {
  return *reinterpret_cast<const v16bf*>(p);
}

__device__ __forceinline__ v8f wmma_bf16(v16bf a, v16bf b, v8f c) {
  return __builtin_amdgcn_wmma_f32_16x16x32_bf16(false, a, false, b, (short)0, c, false, false);
}

__device__ __forceinline__ float fsig(float x)  { return 1.f / (1.f + __expf(-x)); }
__device__ __forceinline__ float ftanh(float x) { float e = __expf(2.f*x); return 1.f - 2.f/(e + 1.f); }

// opaque pointer copy: defeats LICM so weight loads stay inside the time loop
// (otherwise the compiler hoists 512 VGPRs worth of B fragments and spills to scratch)
__device__ __forceinline__ const u16* opaque(const u16* p) {
  asm volatile("" : "+v"(p));
  return p;
}

// ---------------- prep kernels ----------------
__global__ __launch_bounds__(256) void embed_cast(const int* __restrict__ sent,
                                                  const float* __restrict__ emb,
                                                  u16* __restrict__ xb) {
  size_t idx = (size_t)blockIdx.x * 256 + threadIdx.x;      // < T*B*D
  int d  = (int)(idx & (DD - 1));
  int tb = (int)(idx >> 8);
  int v  = sent[tb];
  xb[idx] = f2bf(emb[(size_t)v * DD + d]);
}

__global__ __launch_bounds__(256) void prep_w(const float* __restrict__ wihf, const float* __restrict__ whhf,
                                              const float* __restrict__ wihb, const float* __restrict__ whhb,
                                              u16* __restrict__ wcat) {
  int idx = blockIdx.x * 256 + threadIdx.x;                 // < 2*1024*512 = 2^20
  int dir  = idx >> 19;
  int r    = idx & ((1 << 19) - 1);
  int nrow = r >> 9;
  int k    = r & 511;
  const float* wih = dir ? wihb : wihf;
  const float* whh = dir ? whhb : whhf;
  float v = (k < DD) ? wih[(size_t)nrow * DD + k] : whh[(size_t)nrow * DD + (k - DD)];
  wcat[idx] = f2bf(v);
}

__global__ __launch_bounds__(256) void prep_misc(const float* __restrict__ fcw,
                                                 const float* __restrict__ bihf, const float* __restrict__ bhhf,
                                                 const float* __restrict__ bihb, const float* __restrict__ bhhb,
                                                 u16* __restrict__ fcwp, float* __restrict__ bias) {
  int idx = blockIdx.x * 256 + threadIdx.x;
  if (idx < KKP * 512) {
    int nrow = idx >> 9, k = idx & 511;
    float v = (nrow < KK) ? fcw[(size_t)nrow * 512 + k] : 0.f;
    fcwp[idx] = f2bf(v);
  } else if (idx < KKP * 512 + 2 * GG) {
    int j = idx - KKP * 512;
    int dir = j >> 10, g = j & 1023;
    bias[j] = dir ? (bihb[g] + bhhb[g]) : (bihf[g] + bhhf[g]);
  }
}

// ---------------- BiLSTM: 16 independent workgroups, zero grid syncs ----------------
// Block = (dir, 16-batch block). 16 wave32s; wave w owns d-tile w and all 4 gates for it.
// c-state: VGPR-resident for all 512 steps. h: double-buffered 8KB LDS tile per step.
// Weights stream from L2 (192MB) each step; only sync is one s_barrier per step.
__global__ __launch_bounds__(512) void lstm_wg(const u16* __restrict__ xb,
                                               const u16* __restrict__ wcat,
                                               const float* __restrict__ bias,
                                               u16* __restrict__ hall) {
  __shared__ u16 hbuf[2][16 * DD];       // [parity][m][d]

  const int tid = threadIdx.x;
  const int ln  = tid & 31;
  const int wv  = tid >> 5;              // 0..15  -> d-tile
  const int dir = blockIdx.x >> 3;       // 0: fwd, 1: bwd
  const int bb0 = (blockIdx.x & 7) * 16; // batch block base

  const int lane_lo = (ln < 16);
  const int aoff    = lane_lo ? 0 : 16;  // A-fragment K half
  const int klane   = lane_lo ? 0 : 16;  // B-fragment K half
  const int col     = ln & 15;
  const int d       = wv * 16 + col;     // this lane's d column within its gate tiles
  const int mloc    = lane_lo ? 0 : 8;

  // B row pointers + bias for the 4 gate rows (i,f,g,o) this lane contributes to
  const u16* Bb[4];
  float bsc[4];
#pragma unroll
  for (int g = 0; g < 4; ++g) {
    const int n = g * DD + d;            // gate-major column in G=1024
    Bb[g]  = wcat + ((size_t)dir * GG + n) * 512;
    bsc[g] = bias[(size_t)dir * GG + n];
  }

  // register-resident cell state (C/D-matrix layout: 8 rows per lane)
  float cst[8] = {};
  const v8f zacc = {};

  for (int t = 0; t < TT; ++t) {
    const int tx = dir ? (TT - 1 - t) : t;
    const int rb = t & 1;                // read h from this LDS buffer
    const int wb = rb ^ 1;               // write new h here

    // prefetch next timestep's x tile (wave 0 covers the 8KB block, 256B/lane)
    if (wv == 0 && t + 1 < TT) {
      const int txn = dir ? (tx - 1) : (tx + 1);
      __builtin_prefetch(xb + ((size_t)txn * BB + bb0) * DD + ln * 128, 0, 3);
    }

    // re-materialize weight pointers each step so B streams from L2 (no hoist/spill)
    const u16* Bc[4];
#pragma unroll
    for (int g = 0; g < 4; ++g) Bc[g] = opaque(Bb[g]);

    v8f acc[4];
#pragma unroll
    for (int g = 0; g < 4; ++g) acc[g] = zacc;

    // ---- gate GEMM, x part: K = 0..255 ----
    const u16* xa = xb + ((size_t)tx * BB + bb0 + (ln & 15)) * DD;
#pragma unroll
    for (int kt = 0; kt < 8; ++kt) {
      v16bf a = ldA(xa + kt * 32 + aoff);
#pragma unroll
      for (int g = 0; g < 4; ++g)
        acc[g] = wmma_bf16(a, ldB(Bc[g] + kt * 32 + klane), acc[g]);
    }
    // ---- gate GEMM, h part: K = 256..511 (h_{t-1} from LDS) ----
    if (t > 0) {
      const u16* ha = &hbuf[rb][(ln & 15) * DD];
#pragma unroll
      for (int kt = 0; kt < 8; ++kt) {
        v16bf a = ldA(ha + kt * 32 + aoff);
#pragma unroll
        for (int g = 0; g < 4; ++g)
          acc[g] = wmma_bf16(a, ldB(Bc[g] + DD + kt * 32 + klane), acc[g]);
      }
    }

    // ---- LSTM cell, fully in registers ----
    u16* hw = &hbuf[wb][0];
    u16* hg = hall + (((size_t)dir * TT + t) * BB + bb0) * DD;
#pragma unroll
    for (int r = 0; r < 8; ++r) {
      const float i_ = fsig (acc[0][r] + bsc[0]);
      const float f_ = fsig (acc[1][r] + bsc[1]);
      const float g_ = ftanh(acc[2][r] + bsc[2]);
      const float o_ = fsig (acc[3][r] + bsc[3]);
      const float c  = f_ * cst[r] + i_ * g_;
      cst[r] = c;
      const u16 hv = f2bf(o_ * ftanh(c));
      const int m = r + mloc;
      hw[m * DD + d] = hv;               // LDS: next step's A operand
      hg[(size_t)m * DD + d] = hv;       // global: consumed by FC stage
    }

    __syncthreads();                     // the ONLY per-step synchronization
  }
}

// ---------------- FC projection: feats = [hf|hb] @ fc_w^T + fc_b ----------------
__global__ __launch_bounds__(256) void fc_wmma_kernel(const u16* __restrict__ hall,
                                                      const u16* __restrict__ fcwp,
                                                      const float* __restrict__ fcb,
                                                      float* __restrict__ feats) {
  const int tid   = threadIdx.x;
  const int ln    = tid & 31;
  const int wave  = blockIdx.x * 8 + (tid >> 5);  // 0..8191
  const int ntile = wave & 1;
  const int mtile = wave >> 1;                    // 0..4095
  const int t     = mtile >> 3;                   // B=128 -> 8 m-tiles per t
  const int b0    = (mtile & 7) * 16;
  const int brow  = b0 + (ln & 15);
  const int aoff  = (ln < 16) ? 0 : 16;
  const int n     = ntile * 16 + (ln & 15);
  const int klane = (ln < 16) ? 0 : 16;

  const u16* wr = fcwp + (size_t)n * 512;
  const u16* hf = hall + ((size_t)t * BB + brow) * DD;                         // dir 0
  const u16* hb = hall + (((size_t)TT + (TT - 1 - t)) * BB + brow) * DD;       // dir 1 (reversed order)

  v8f acc = {};
#pragma unroll
  for (int kt = 0; kt < 8; ++kt)
    acc = wmma_bf16(ldA(hf + kt * 32 + aoff), ldB(wr + kt * 32 + klane), acc);
#pragma unroll
  for (int kt = 0; kt < 8; ++kt)
    acc = wmma_bf16(ldA(hb + kt * 32 + aoff), ldB(wr + DD + kt * 32 + klane), acc);

  const int col = ntile * 16 + (ln & 15);
  if (col < KK) {
    const float bv = fcb[col];
    const int rbase = b0 + ((ln < 16) ? 0 : 8);
#pragma unroll
    for (int r = 0; r < 8; ++r)
      feats[((size_t)t * BB + rbase + r) * KK + col] = acc[r] + bv;
  }
}

// ---------------- CRF: one wave per batch chain ----------------
__global__ __launch_bounds__(32) void crf_kernel(const float* __restrict__ feats,
                                                 const int* __restrict__ tags,
                                                 const float* __restrict__ start,
                                                 const float* __restrict__ endv,
                                                 const float* __restrict__ trans,
                                                 float* __restrict__ llh) {
  const int b = blockIdx.x;
  const int j = threadIdx.x;
  __shared__ float tr[KK * KK];
  __shared__ float sc[KK];
  __shared__ float red[32];
  for (int i = j; i < KK * KK; i += 32) tr[i] = trans[i];
  __syncthreads();

  float score = 0.f;
  if (j < KK) score = start[j] + feats[(size_t)b * KK + j];
  for (int t = 1; t < TT; ++t) {
    if (j < KK) sc[j] = score;
    __syncthreads();
    if (j < KK) {
      float m = -1e30f;
#pragma unroll
      for (int i = 0; i < KK; ++i) m = fmaxf(m, sc[i] + tr[i * KK + j]);
      float s = 0.f;
#pragma unroll
      for (int i = 0; i < KK; ++i) s += __expf(sc[i] + tr[i * KK + j] - m);
      score = m + __logf(s) + feats[((size_t)t * BB + b) * KK + j];
    }
    __syncthreads();
  }
  if (j < KK) sc[j] = score + endv[j];
  __syncthreads();

  float part = 0.f;
  for (int t = 1 + j; t < TT; t += 32) {
    const int yt = tags[(size_t)t * BB + b];
    const int yp = tags[(size_t)(t - 1) * BB + b];
    part += tr[yp * KK + yt] + feats[((size_t)t * BB + b) * KK + yt];
  }
  red[j] = part;
  __syncthreads();

  if (j == 0) {
    float m = -1e30f;
    for (int i = 0; i < KK; ++i) m = fmaxf(m, sc[i]);
    float s = 0.f;
    for (int i = 0; i < KK; ++i) s += __expf(sc[i] - m);
    const float denom = m + __logf(s);
    float num = 0.f;
    for (int i = 0; i < 32; ++i) num += red[i];
    const int y0 = tags[b];
    const int yl = tags[(size_t)(TT - 1) * BB + b];
    num += start[y0] + feats[(size_t)b * KK + y0] + endv[yl];
    llh[b] = num - denom;
  }
}

__global__ __launch_bounds__(128) void reduce_llh(const float* __restrict__ llh, float* __restrict__ out) {
  __shared__ float s[128];
  s[threadIdx.x] = llh[threadIdx.x];
  __syncthreads();
  for (int st = 64; st > 0; st >>= 1) {
    if (threadIdx.x < st) s[threadIdx.x] += s[threadIdx.x + st];
    __syncthreads();
  }
  if (threadIdx.x == 0) out[0] = s[0];
}

// ---------------- launch ----------------
extern "C" void kernel_launch(void* const* d_in, const int* in_sizes, int n_in,
                              void* d_out, int out_size, void* d_ws, size_t ws_size,
                              hipStream_t stream) {
  const int*   sentences = (const int*)  d_in[0];
  const int*   tags      = (const int*)  d_in[1];
  const float* emb       = (const float*)d_in[2];
  const float* w_ih_f    = (const float*)d_in[3];
  const float* w_hh_f    = (const float*)d_in[4];
  const float* b_ih_f    = (const float*)d_in[5];
  const float* b_hh_f    = (const float*)d_in[6];
  const float* w_ih_b    = (const float*)d_in[7];
  const float* w_hh_b    = (const float*)d_in[8];
  const float* b_ih_b    = (const float*)d_in[9];
  const float* b_hh_b    = (const float*)d_in[10];
  const float* fc_w      = (const float*)d_in[11];
  const float* fc_b      = (const float*)d_in[12];
  const float* crf_start = (const float*)d_in[13];
  const float* crf_end   = (const float*)d_in[14];
  const float* crf_trans = (const float*)d_in[15];

  char* ws = (char*)d_ws;
  u16*   xb    = (u16*)  (ws + OFF_X);
  u16*   wcat  = (u16*)  (ws + OFF_WCAT);
  float* bias  = (float*)(ws + OFF_BIAS);
  u16*   fcwp  = (u16*)  (ws + OFF_FCW);
  u16*   hall  = (u16*)  (ws + OFF_HALL);
  float* feats = (float*)(ws + OFF_FEAT);
  float* llh   = (float*)(ws + OFF_LLH);

  embed_cast<<<(TT * BB * DD) / 256, 256, 0, stream>>>(sentences, emb, xb);
  prep_w<<<(2 * GG * 512) / 256, 256, 0, stream>>>(w_ih_f, w_hh_f, w_ih_b, w_hh_b, wcat);
  prep_misc<<<(KKP * 512 + 2 * GG) / 256, 256, 0, stream>>>(
      fc_w, b_ih_f, b_hh_f, b_ih_b, b_hh_b, fcwp, bias);
  lstm_wg<<<16, 512, 0, stream>>>(xb, wcat, bias, hall);
  fc_wmma_kernel<<<1024, 256, 0, stream>>>(hall, fcwp, fc_b, feats);
  crf_kernel<<<BB, 32, 0, stream>>>(feats, tags, crf_start, crf_end, crf_trans, llh);
  reduce_llh<<<1, 128, 0, stream>>>(llh, (float*)d_out);
}
